// CausalSelfAttention_24635932409993
// MI455X (gfx1250) — matrix operand order repaired and checked by reading, not compiled, not run on hardware
//
#include <hip/hip_runtime.h>
#include <hip/hip_bf16.h>

typedef __attribute__((ext_vector_type(16))) _Float16 v16h;
typedef __attribute__((ext_vector_type(8)))  float    v8f;

// ---------------------------------------------------------------------------
// WMMA fragment loaders from LDS (contiguous k order k = 16*(L>>4) + s, wave32)
// ---------------------------------------------------------------------------
// A-matrix 16x32 f16: lane L -> row m=L&15; lanes 16-31 K-offset +16.
// VGPR i holds K pair: k = 2*i + koff  (pair is contiguous).
__device__ __forceinline__ v16h frag_a_lds(const _Float16* base, int lda_h) {
  const int lane = threadIdx.x & 31;
  const int m    = lane & 15;
  const int koff = (lane >> 4) << 4;            // 0 or 16
  union { v16h v; unsigned u[8]; } r;
  const unsigned* row = (const unsigned*)(base + m * lda_h);
#pragma unroll
  for (int i = 0; i < 8; ++i) {
    const int k = (i << 1) + koff;  // even
    r.u[i] = row[k >> 1];
  }
  return r.v;
}

// B-matrix 32x16 f16, staged n-major in LDS (Bt[n][k]):
// lane L -> col n=L&15; lanes 0-15 hold K=0..15, lanes 16-31 hold K=16..31.
__device__ __forceinline__ v16h frag_b_lds(const _Float16* baseT, int ldb_h) {
  const int lane = threadIdx.x & 31;
  const int n    = lane & 15;
  const int koff = (lane >> 4) << 4;            // 0 or 16
  union { v16h v; unsigned u[8]; } r;
  const unsigned* row = (const unsigned*)(baseT + n * ldb_h + koff);
#pragma unroll
  for (int i = 0; i < 8; ++i) r.u[i] = row[i];
  return r.v;
}
// C/D 16x16 f32: lane L holds col n=L&15, VGPR i holds row m = i + 8*(L>>4).

// log2(10000)/64
#define ROPE_L2T_OVER_D 0.207620506f

// ---------------------------------------------------------------------------
// Kernel 1: qkv = x @ Wqkv + bqkv, split + RoPE, store f16 in [B,H,T,D]
// M=8192, N=3072, K=1024. Block: 256 thr (8 waves), tile 128M x 64N, Kstep 32.
// ---------------------------------------------------------------------------
#define LDA 40
#define LDB 40

__global__ void __launch_bounds__(256)
qkv_rope_kernel(const float* __restrict__ x, const float* __restrict__ W,
                const float* __restrict__ bias,
                _Float16* __restrict__ qf, _Float16* __restrict__ kf,
                _Float16* __restrict__ vf) {
  __shared__ _Float16 As[128 * LDA];
  __shared__ _Float16 Bs[64 * LDB];
  const int K = 1024, N = 3072, T = 2048, H = 16, D = 64;
  const int m_base = blockIdx.y * 128;
  const int n_base = blockIdx.x * 64;
  const int tid = threadIdx.x, lane = tid & 31;
  const int w  = tid >> 5;
  const int wm = w >> 1, wn = w & 1;            // 4x2 wave grid

  v8f acc[2][2] = {};

  for (int k0 = 0; k0 < K; k0 += 32) {
    // stage A: 128x32 f32 -> f16 (each thread: 16 contiguous elems)
    {
      const int row = tid >> 1;
      const int col = (tid & 1) << 4;
      const float* src = x + (size_t)(m_base + row) * K + k0 + col;
      if (k0 + 32 < K) __builtin_prefetch(src + 32, 0, 1);
      _Float16* dst = As + row * LDA + col;
#pragma unroll
      for (int j = 0; j < 16; ++j) dst[j] = (_Float16)src[j];
    }
    // stage B transposed: W[k][n] -> Bs[n][k], 32x64
    {
      const int kk = tid & 31;
      const int n8 = (tid >> 5) << 3;
      const float* src = W + (size_t)(k0 + kk) * N + n_base + n8;
#pragma unroll
      for (int j = 0; j < 8; ++j) Bs[(n8 + j) * LDB + kk] = (_Float16)src[j];
    }
    __syncthreads();
    {
      v16h a0 = frag_a_lds(As + (wm * 32 +  0) * LDA, LDA);
      v16h a1 = frag_a_lds(As + (wm * 32 + 16) * LDA, LDA);
      v16h b0 = frag_b_lds(Bs + (wn * 32 +  0) * LDB, LDB);
      v16h b1 = frag_b_lds(Bs + (wn * 32 + 16) * LDB, LDB);
      acc[0][0] = __builtin_amdgcn_wmma_f32_16x16x32_f16(false, a0, false, b0,
                                                         (short)0, acc[0][0], false, false);
      acc[0][1] = __builtin_amdgcn_wmma_f32_16x16x32_f16(false, a0, false, b1,
                                                         (short)0, acc[0][1], false, false);
      acc[1][0] = __builtin_amdgcn_wmma_f32_16x16x32_f16(false, a1, false, b0,
                                                         (short)0, acc[1][0], false, false);
      acc[1][1] = __builtin_amdgcn_wmma_f32_16x16x32_f16(false, a1, false, b1,
                                                         (short)0, acc[1][1], false, false);
    }
    __syncthreads();
  }

  // epilogue: bias + RoPE (pair partner is lane^1) + scatter to [B,H,T,D] f16
  const int half = lane >> 4, ncol = lane & 15;
#pragma unroll
  for (int ti = 0; ti < 2; ++ti) {
#pragma unroll
    for (int tj = 0; tj < 2; ++tj) {
      const int n   = n_base + wn * 32 + tj * 16 + ncol;
      const int sec = n >> 10;                  // 0=q 1=k 2=v (uniform per tile)
      const int nl  = n & 1023;
      const int h   = nl >> 6;
      const int d   = nl & 63;
      const float bv = bias[n];
      // inv_freq = theta^(-(d&~1)/D) = exp2(-(d&~1) * log2(theta)/D)
      const float invf = exp2f(-(float)(d & ~1) * ROPE_L2T_OVER_D);
#pragma unroll
      for (int i = 0; i < 8; ++i) {
        const int m = m_base + wm * 32 + ti * 16 + i + (half << 3);
        const int t = m & (T - 1);
        const int b = m >> 11;
        float val = acc[ti][tj][i] + bv;
        float out;
        if (sec < 2) {
          const float partner = __shfl_xor(val, 1, 32);
          const float ang = (float)t * invf;
          const float c = __cosf(ang), s = __sinf(ang);
          out = (d & 1) ? (partner * s + val * c) : (val * c - partner * s);
        } else {
          out = val;
        }
        const size_t idx = ((size_t)(b * H + h) * T + t) * D + d;
        if (sec == 0)      qf[idx] = (_Float16)out;
        else if (sec == 1) kf[idx] = (_Float16)out;
        else               vf[idx] = (_Float16)out;
      }
    }
  }
}

// ---------------------------------------------------------------------------
// Kernel 2: causal flash attention. Block = (b,h) x 128-query tile, 8 waves,
// each wave owns 16 q-rows x 64-key chunk. Online softmax, f32 accum.
// ---------------------------------------------------------------------------
#define LQ 72

__global__ void __launch_bounds__(256)
attn_kernel(const _Float16* __restrict__ qf, const _Float16* __restrict__ kf,
            const _Float16* __restrict__ vf, _Float16* __restrict__ attn) {
  __shared__ _Float16 Qs[128 * LQ];             // q-tile [qrow][d]
  __shared__ _Float16 Ks[64 * LQ];              // k-chunk [key][d]  (== Bt for QK^T)
  __shared__ _Float16 Vt[64 * LQ];              // v-chunk [d][key]  (== Bt for PV)
  __shared__ _Float16 Ps[8 * 16 * LQ];          // per-wave P strip [qrow][key]
  const int T = 2048, D = 64;
  const int q0 = blockIdx.x * 128;
  const int bh = blockIdx.y;
  const size_t base = (size_t)bh * T * D;
  const int tid = threadIdx.x, lane = tid & 31, w = tid >> 5;
  const int half = lane >> 4, ncol = lane & 15;

  // load Q tile (32 contiguous halfs per thread)
  {
    const int row = tid >> 1;
    const int col = (tid & 1) << 5;
    const _Float16* src = qf + base + (size_t)(q0 + row) * D + col;
    _Float16* dst = Qs + row * LQ + col;
#pragma unroll
    for (int j = 0; j < 32; ++j) dst[j] = src[j];
  }

  v8f O[4] = {};
  float mrow[8], lrow[8];
#pragma unroll
  for (int i = 0; i < 8; ++i) { mrow[i] = -3.0e38f; lrow[i] = 0.0f; }

  const int nChunks = (q0 >> 6) + 2;            // causal: keys <= q0+127
  for (int kc = 0; kc < nChunks; ++kc) {
    __syncthreads();
    // stage K chunk row-major [key][d]
    {
      const int row = tid >> 2;
      const int col = (tid & 3) << 4;
      const _Float16* src = kf + base + (size_t)(kc * 64 + row) * D + col;
      _Float16* dst = Ks + row * LQ + col;
#pragma unroll
      for (int j = 0; j < 16; ++j) dst[j] = src[j];
    }
    // stage V chunk transposed Vt[d][key]
    {
      const int key = tid >> 2;
      const int col = (tid & 3) << 4;
      const _Float16* src = vf + base + (size_t)(kc * 64 + key) * D + col;
#pragma unroll
      for (int j = 0; j < 16; ++j) Vt[(col + j) * LQ + key] = src[j];
    }
    __syncthreads();

    // S = Q K^T  (16 q-rows x 64 keys per wave; D=64 -> 2 k-steps)
    v8f S[4];
    {
      v16h aq0 = frag_a_lds(Qs + (w * 16) * LQ +  0, LQ);
      v16h aq1 = frag_a_lds(Qs + (w * 16) * LQ + 32, LQ);
#pragma unroll
      for (int tj = 0; tj < 4; ++tj) {
        v16h b0 = frag_b_lds(Ks + (tj * 16) * LQ +  0, LQ);
        v16h b1 = frag_b_lds(Ks + (tj * 16) * LQ + 32, LQ);
        v8f s = {};
        s = __builtin_amdgcn_wmma_f32_16x16x32_f16(false, aq0, false, b0,
                                                   (short)0, s, false, false);
        s = __builtin_amdgcn_wmma_f32_16x16x32_f16(false, aq1, false, b1,
                                                   (short)0, s, false, false);
        S[tj] = s;
      }
    }

    // scale, causal mask, online softmax (row lives in 16 lanes of half-wave)
#pragma unroll
    for (int i = 0; i < 8; ++i) {
      const int qrow = q0 + w * 16 + i + (half << 3);
      float mx = mrow[i];
#pragma unroll
      for (int tj = 0; tj < 4; ++tj) {
        const int kcol = kc * 64 + tj * 16 + ncol;
        float v = S[tj][i] * 0.125f;
        if (kcol > qrow) v = -3.0e38f;
        S[tj][i] = v;
        mx = fmaxf(mx, v);
      }
#pragma unroll
      for (int off = 1; off < 16; off <<= 1) mx = fmaxf(mx, __shfl_xor(mx, off, 32));
      const float alpha = __expf(mrow[i] - mx);
      float rsum = 0.0f;
#pragma unroll
      for (int tj = 0; tj < 4; ++tj) {
        const float p = __expf(S[tj][i] - mx);
        S[tj][i] = p;
        rsum += p;
      }
#pragma unroll
      for (int off = 1; off < 16; off <<= 1) rsum += __shfl_xor(rsum, off, 32);
      lrow[i] = lrow[i] * alpha + rsum;
      mrow[i] = mx;
#pragma unroll
      for (int tn = 0; tn < 4; ++tn) O[tn][i] *= alpha;
    }

    // C-layout P -> A-layout via wave-private LDS strip (DS ops are in-order)
    _Float16* Pw = Ps + w * 16 * LQ;
#pragma unroll
    for (int tj = 0; tj < 4; ++tj)
#pragma unroll
      for (int i = 0; i < 8; ++i)
        Pw[(i + (half << 3)) * LQ + tj * 16 + ncol] = (_Float16)S[tj][i];

    // O += P @ V
#pragma unroll
    for (int ks = 0; ks < 2; ++ks) {
      v16h a = frag_a_lds(Pw + ks * 32, LQ);
#pragma unroll
      for (int tn = 0; tn < 4; ++tn) {
        v16h b = frag_b_lds(Vt + (tn * 16) * LQ + ks * 32, LQ);
        O[tn] = __builtin_amdgcn_wmma_f32_16x16x32_f16(false, a, false, b,
                                                       (short)0, O[tn], false, false);
      }
    }
  }

  // normalize and store to attn f16 in [B, T, H*D] (= [B,T,E] row-major)
  const int b = bh >> 4, h = bh & 15;
#pragma unroll
  for (int i = 0; i < 8; ++i) {
    const float inv = 1.0f / lrow[i];
    const int t = q0 + w * 16 + i + (half << 3);
    const size_t rowoff = (size_t)(b * 2048 + t) * 1024 + h * 64;
#pragma unroll
    for (int tn = 0; tn < 4; ++tn)
      attn[rowoff + tn * 16 + ncol] = (_Float16)(O[tn][i] * inv);
  }
}

// ---------------------------------------------------------------------------
// Kernel 3: out = attn @ Wproj + bproj. M=8192, N=K=1024. Same tiling.
// ---------------------------------------------------------------------------
__global__ void __launch_bounds__(256)
proj_kernel(const _Float16* __restrict__ A, const float* __restrict__ W,
            const float* __restrict__ bias, float* __restrict__ out) {
  __shared__ _Float16 As[128 * LDA];
  __shared__ _Float16 Bs[64 * LDB];
  const int K = 1024, N = 1024;
  const int m_base = blockIdx.y * 128;
  const int n_base = blockIdx.x * 64;
  const int tid = threadIdx.x, lane = tid & 31;
  const int w  = tid >> 5;
  const int wm = w >> 1, wn = w & 1;

  v8f acc[2][2] = {};

  for (int k0 = 0; k0 < K; k0 += 32) {
    {
      const int row = tid >> 1;
      const int col = (tid & 1) << 4;
      const _Float16* src = A + (size_t)(m_base + row) * K + k0 + col;
      if (k0 + 32 < K) __builtin_prefetch(src + 32, 0, 1);
      _Float16* dst = As + row * LDA + col;
#pragma unroll
      for (int j = 0; j < 16; ++j) dst[j] = src[j];
    }
    {
      const int kk = tid & 31;
      const int n8 = (tid >> 5) << 3;
      const float* src = W + (size_t)(k0 + kk) * N + n_base + n8;
#pragma unroll
      for (int j = 0; j < 8; ++j) Bs[(n8 + j) * LDB + kk] = (_Float16)src[j];
    }
    __syncthreads();
    {
      v16h a0 = frag_a_lds(As + (wm * 32 +  0) * LDA, LDA);
      v16h a1 = frag_a_lds(As + (wm * 32 + 16) * LDA, LDA);
      v16h b0 = frag_b_lds(Bs + (wn * 32 +  0) * LDB, LDB);
      v16h b1 = frag_b_lds(Bs + (wn * 32 + 16) * LDB, LDB);
      acc[0][0] = __builtin_amdgcn_wmma_f32_16x16x32_f16(false, a0, false, b0,
                                                         (short)0, acc[0][0], false, false);
      acc[0][1] = __builtin_amdgcn_wmma_f32_16x16x32_f16(false, a0, false, b1,
                                                         (short)0, acc[0][1], false, false);
      acc[1][0] = __builtin_amdgcn_wmma_f32_16x16x32_f16(false, a1, false, b0,
                                                         (short)0, acc[1][0], false, false);
      acc[1][1] = __builtin_amdgcn_wmma_f32_16x16x32_f16(false, a1, false, b1,
                                                         (short)0, acc[1][1], false, false);
    }
    __syncthreads();
  }

  const int half = lane >> 4, ncol = lane & 15;
#pragma unroll
  for (int ti = 0; ti < 2; ++ti)
#pragma unroll
    for (int tj = 0; tj < 2; ++tj) {
      const int n = n_base + wn * 32 + tj * 16 + ncol;
      const float bv = bias[n];
#pragma unroll
      for (int i = 0; i < 8; ++i) {
        const int m = m_base + wm * 32 + ti * 16 + i + (half << 3);
        out[(size_t)m * N + n] = acc[ti][tj][i] + bv;
      }
    }
}

// ---------------------------------------------------------------------------
extern "C" void kernel_launch(void* const* d_in, const int* in_sizes, int n_in,
                              void* d_out, int out_size, void* d_ws, size_t ws_size,
                              hipStream_t stream) {
  const float* x     = (const float*)d_in[0];
  const float* Wqkv  = (const float*)d_in[1];
  const float* bqkv  = (const float*)d_in[2];
  const float* Wproj = (const float*)d_in[3];
  const float* bproj = (const float*)d_in[4];
  float* out = (float*)d_out;

  const size_t qkv_elems = (size_t)4 * 16 * 2048 * 64;   // 8.39M halfs each
  _Float16* qf   = (_Float16*)d_ws;
  _Float16* kf   = qf + qkv_elems;
  _Float16* vf   = kf + qkv_elems;
  _Float16* attn = vf + qkv_elems;                       // 8192*1024 halfs

  dim3 g1(3072 / 64, 8192 / 128);   // (48, 64)
  qkv_rope_kernel<<<g1, 256, 0, stream>>>(x, Wqkv, bqkv, qf, kf, vf);

  dim3 g2(2048 / 128, 64);          // (16, B*H)
  attn_kernel<<<g2, 256, 0, stream>>>(qf, kf, vf, attn);

  dim3 g3(1024 / 64, 8192 / 128);   // (16, 64)
  proj_kernel<<<g3, 256, 0, stream>>>(attn, Wproj, bproj, out);
}
